// PointNetConv_88553635709089
// MI455X (gfx1250) — matrix operand order repaired and checked by reading, not compiled, not run on hardware
//
#include <hip/hip_runtime.h>
#include <hip/hip_bf16.h>

typedef __attribute__((ext_vector_type(16))) _Float16 v16h;
typedef __attribute__((ext_vector_type(8)))  float    v8f;

#define D_FEAT 64
#define P_POS  3
#define K1PAD  96      // 67 padded up to 3 * 32
#define HDIM   128
#define ODIM   128
#define TILE_E 128     // edges per block (8 waves x 16 edges)

union AFrag { v16h v; float4 q[2]; };
union BFrag { v16h v; float4 q[2]; };

// ---------------- small helper kernels ----------------

__global__ void init_out_kernel(float* out, int n) {
  int i = blockIdx.x * blockDim.x + threadIdx.x;
  if (i < n) out[i] = -__builtin_inff();
}

__global__ void finalize_out_kernel(float* out, int n) {
  int i = blockIdx.x * blockDim.x + threadIdx.x;
  if (i < n) {
    float v = out[i];
    out[i] = (v == -__builtin_inff()) ? 0.f : v;
  }
}

// Transpose + f16-convert weights:  W1t[n][k] (k in [0,96)), W2t[n][k] (k in [0,128))
__global__ void prep_weights_kernel(const float* __restrict__ W1,
                                    const float* __restrict__ W2,
                                    _Float16* __restrict__ W1t,
                                    _Float16* __restrict__ W2t) {
  int i = blockIdx.x * blockDim.x + threadIdx.x;
  if (i < HDIM * K1PAD) {
    int n = i / K1PAD, k = i % K1PAD;
    W1t[i] = (k < D_FEAT + P_POS) ? (_Float16)W1[k * HDIM + n] : (_Float16)0.f;
  } else if (i < HDIM * K1PAD + ODIM * HDIM) {
    int j = i - HDIM * K1PAD;
    int n = j / HDIM, k = j % HDIM;
    W2t[j] = (_Float16)W2[k * ODIM + n];
  }
}

// ---------------- main fused kernel ----------------
// One wave = one 16-edge M-tile. Wave-private LDS region of 16*128 f16 (4KB);
// no cross-wave sharing -> no barriers.

__global__ __launch_bounds__(256)
void edge_mlp_scatter_max_kernel(const float* __restrict__ x,
                                 const float* __restrict__ pos,
                                 const float* __restrict__ pos_c,
                                 const int*   __restrict__ src,
                                 const int*   __restrict__ dst,
                                 const _Float16* __restrict__ W1t,
                                 const float* __restrict__ b1,
                                 const _Float16* __restrict__ W2t,
                                 const float* __restrict__ b2,
                                 float* __restrict__ out,
                                 int E) {
  __shared__ _Float16 sm[8 * 16 * HDIM];           // 32 KB
  const int tid  = threadIdx.x;
  const int wave = tid >> 5;
  const int lane = tid & 31;
  const int col  = lane & 15;                      // A-row / B-col / C-col index
  const int hsel = lane >> 4;                      // lane-half select
  const int mbase = hsel * 8;                      // C/D rows this lane-half holds

  _Float16* my = sm + wave * (16 * HDIM);
  const char* myb = reinterpret_cast<const char*>(my);
  const int eBase = blockIdx.x * TILE_E + wave * 16;

  // ---- Phase A: gather x[src] as f16 into LDS rows of stride K1PAD ----
  for (int i = lane; i < 16 * 16; i += 32) {       // 16 edges x 16 float4 each
    int e  = i >> 4;
    int k4 = i & 15;
    int ge = eBase + e; if (ge >= E) ge = E - 1;   // clamp (idempotent under max)
    int se = src[ge];
    float4 vv = reinterpret_cast<const float4*>(x)[se * 16 + k4];
    union { _Float16 h[4]; unsigned long long u; } t;
    t.h[0] = (_Float16)vv.x; t.h[1] = (_Float16)vv.y;
    t.h[2] = (_Float16)vv.z; t.h[3] = (_Float16)vv.w;
    *reinterpret_cast<unsigned long long*>(my + e * K1PAD + k4 * 4) = t.u;
  }

  // ---- Phase B: relative position + zero padding (lanes 0..15, 1 edge each) ----
  if (lane < 16) {
    int e  = lane;
    int ge = eBase + e; if (ge >= E) ge = E - 1;
    int se = src[ge], de = dst[ge];
    float d0 = pos[se * 3 + 0] - pos_c[de * 3 + 0];
    float d1 = pos[se * 3 + 1] - pos_c[de * 3 + 1];
    float d2 = pos[se * 3 + 2] - pos_c[de * 3 + 2];
    _Float16* row = my + e * K1PAD;
    row[64] = (_Float16)d0; row[65] = (_Float16)d1; row[66] = (_Float16)d2;
    for (int k = 67; k < K1PAD; ++k) row[k] = (_Float16)0.f;
  }

  // ---- Phase C: GEMM1  [16 x 96] @ [96 x 128] -> f32 acc ----
  v8f acc[8] = {};
#pragma unroll
  for (int kt = 0; kt < 3; ++kt) {
    AFrag a;
    int aoff = col * (K1PAD * 2) + kt * 64 + hsel * 16;   // bytes
    a.q[0] = *reinterpret_cast<const float4*>(myb + aoff);
    a.q[1] = *reinterpret_cast<const float4*>(myb + aoff + 32);
#pragma unroll
    for (int nt = 0; nt < 8; ++nt) {
      BFrag b;
      const char* wb = reinterpret_cast<const char*>(W1t) +
                       (nt * 16 + col) * (K1PAD * 2) + kt * 64 + hsel * 16;
      b.q[0] = *reinterpret_cast<const float4*>(wb);
      b.q[1] = *reinterpret_cast<const float4*>(wb + 32);
      acc[nt] = __builtin_amdgcn_wmma_f32_16x16x32_f16(
          false, a.v, false, b.v, (short)0, acc[nt], false, false);
    }
  }

  // ---- Phase D: bias + ReLU, write hidden back to LDS as f16 [16][128] ----
#pragma unroll
  for (int nt = 0; nt < 8; ++nt) {
    int c = nt * 16 + col;
    float bias = b1[c];
#pragma unroll
    for (int r = 0; r < 8; ++r) {
      float v = acc[nt][r] + bias;
      v = v > 0.f ? v : 0.f;
      my[(mbase + r) * HDIM + c] = (_Float16)v;     // same-wave LDS: HW in-order
    }
  }

  // ---- Phase E: GEMM2  [16 x 128] @ [128 x 128] -> f32 acc2 ----
  v8f acc2[8] = {};
#pragma unroll
  for (int kt = 0; kt < 4; ++kt) {
    AFrag a;
    int aoff = col * (HDIM * 2) + kt * 64 + hsel * 16;    // bytes
    a.q[0] = *reinterpret_cast<const float4*>(myb + aoff);
    a.q[1] = *reinterpret_cast<const float4*>(myb + aoff + 32);
#pragma unroll
    for (int nt = 0; nt < 8; ++nt) {
      BFrag b;
      const char* wb = reinterpret_cast<const char*>(W2t) +
                       (nt * 16 + col) * (HDIM * 2) + kt * 64 + hsel * 16;
      b.q[0] = *reinterpret_cast<const float4*>(wb);
      b.q[1] = *reinterpret_cast<const float4*>(wb + 32);
      acc2[nt] = __builtin_amdgcn_wmma_f32_16x16x32_f16(
          false, a.v, false, b.v, (short)0, acc2[nt], false, false);
    }
  }

  // ---- Phase F: bias + scatter-max onto centers ----
  int myDst[8];
#pragma unroll
  for (int r = 0; r < 8; ++r) {
    int e = eBase + mbase + r;
    myDst[r] = (e < E) ? dst[e] : -1;
  }
#pragma unroll
  for (int nt = 0; nt < 8; ++nt) {
    int c = nt * 16 + col;
    float bias = b2[c];
#pragma unroll
    for (int r = 0; r < 8; ++r) {
      int d = myDst[r];
      if (d >= 0) {
        float v = acc2[nt][r] + bias;
        __hip_atomic_fetch_max(&out[d * ODIM + c], v,
                               __ATOMIC_RELAXED, __HIP_MEMORY_SCOPE_AGENT);
      }
    }
  }
}

// ---------------- launcher ----------------

extern "C" void kernel_launch(void* const* d_in, const int* in_sizes, int n_in,
                              void* d_out, int out_size, void* d_ws, size_t ws_size,
                              hipStream_t stream) {
  const float* x     = (const float*)d_in[0];
  // d_in[1] (x_c) only used for shape on host side
  const float* pos   = (const float*)d_in[2];
  const float* pos_c = (const float*)d_in[3];
  const int*   src   = (const int*)d_in[4];
  const int*   dst   = (const int*)d_in[5];
  const float* W1    = (const float*)d_in[6];
  const float* b1    = (const float*)d_in[7];
  const float* W2    = (const float*)d_in[8];
  const float* b2    = (const float*)d_in[9];
  float* out = (float*)d_out;
  const int E = in_sizes[4];

  _Float16* W1t = (_Float16*)d_ws;                 // 128*96  f16 = 24 KB
  _Float16* W2t = W1t + HDIM * K1PAD;              // 128*128 f16 = 32 KB

  int prep_total = HDIM * K1PAD + ODIM * HDIM;
  prep_weights_kernel<<<(prep_total + 255) / 256, 256, 0, stream>>>(W1, W2, W1t, W2t);
  init_out_kernel<<<(out_size + 255) / 256, 256, 0, stream>>>(out, out_size);

  int nblocks = (E + TILE_E - 1) / TILE_E;
  edge_mlp_scatter_max_kernel<<<nblocks, 256, 0, stream>>>(
      x, pos, pos_c, src, dst, W1t, b1, W2t, b2, out, E);

  finalize_out_kernel<<<(out_size + 255) / 256, 256, 0, stream>>>(out, out_size);
}
